// GenericHRNN_50723563766322
// MI455X (gfx1250) — matrix operand order repaired and checked by reading, not compiled, not run on hardware
//
#include <hip/hip_runtime.h>
#include <hip/hip_bf16.h>

typedef __attribute__((ext_vector_type(16))) _Float16 v16h;
typedef __attribute__((ext_vector_type(8)))  _Float16 v8h;
typedef __attribute__((ext_vector_type(8)))  float    v8f;

#define L_   3
#define H_   512
#define IN_  512
#define B_   64
#define T_   1024
#define NWG  8
#define HS   (H_ / NWG)     // 64 columns of H per workgroup
#define KT_H (H_ / 32)      // 16 k-tiles for K=512
#define NT_G (3 * H_ / 16)  // 96 n-tiles for 3H gate dim
#define NT_H (H_ / 16)      // 32 n-tiles for H

#define WTILES      24                 // 12 Wih + 12 Whh 1KB tiles staged per k-step
#define WBUF_HALFS  (WTILES * 512)     // 24KB per staging buffer
#define SMEM_BYTES  (B_ * IN_ * 2 + 2 * WBUF_HALFS * 2)   // 64KB x_t + 2x24KB weights = 112KB

#if defined(__has_builtin)
#if __has_builtin(__builtin_amdgcn_global_load_async_to_lds_b128) && \
    __has_builtin(__builtin_amdgcn_s_wait_asynccnt)
#define HAVE_ASYNC_LDS 1
#endif
#endif

#ifdef HAVE_ASYNC_LDS
typedef __attribute__((ext_vector_type(4))) int v4i;
typedef __attribute__((address_space(1))) v4i* gas_v4ip;   // global int4*
typedef __attribute__((address_space(3))) v4i* las_v4ip;   // LDS int4*
#endif

// ---------------- helpers ----------------

__device__ __forceinline__ v8f vzero8() {
  v8f v;
  for (int i = 0; i < 8; ++i) v[i] = 0.0f;
  return v;
}

__device__ __forceinline__ float sigmoidf_(float x) {
  return 1.0f / (1.0f + __expf(-x));
}

// A-fragment (16x32 f16, M=lane&15): lane<16 holds K {0..7,16..23}, lane>=16 holds K {8..15,24..31}
__device__ __forceinline__ v16h load_afrag(const _Float16* A, int ld, int m0, int k0, int lane) {
  const int lm = lane & 15, lh = lane >> 4;
  const _Float16* p = A + (size_t)(m0 + lm) * ld + k0 + lh * 8;
  union { v16h v; v8h h[2]; } u;
  u.h[0] = *(const v8h*)(p);
  u.h[1] = *(const v8h*)(p + 16);
  return u.v;
}

// B-fragment from pre-swizzled weights (global): 32B contiguous per lane
__device__ __forceinline__ v16h load_bfrag(const _Float16* Bs, int kt_total, int ntile, int ktile, int lane) {
  return *(const v16h*)(Bs + (((size_t)ntile * kt_total + ktile) * 32 + lane) * 16);
}

__device__ __forceinline__ v8f wmma16(v16h a, v16h b, v8f c) {
  return __builtin_amdgcn_wmma_f32_16x16x32_f16(false, a, false, b, (short)0, c, false, false);
}

// device-scope barrier (monotonic counter)
__device__ __forceinline__ void gbar(unsigned* cnt, unsigned& gen) {
  __threadfence();
  __syncthreads();
  gen++;
  if (threadIdx.x == 0) {
    __hip_atomic_fetch_add(cnt, 1u, __ATOMIC_ACQ_REL, __HIP_MEMORY_SCOPE_AGENT);
    const unsigned target = gen * NWG;
    while (__hip_atomic_load(cnt, __ATOMIC_ACQUIRE, __HIP_MEMORY_SCOPE_AGENT) < target) {
      __builtin_amdgcn_s_sleep(2);
    }
  }
  __syncthreads();
}

// stage the 24 x 1KB weight tiles this WG needs for k-step `kt` into LDS
// (async DMA global->LDS when available; tracked with ASYNCcnt)
__device__ __forceinline__ void stage_wtiles(const _Float16* __restrict__ Wi,
                                             const _Float16* __restrict__ Wh,
                                             _Float16* dst, int wg, int kt, int tid) {
#pragma unroll
  for (int j = 0; j < 6; ++j) {
    const int idx = tid + j * 256;     // 1536 chunks of 16B = 24KB
    const int tile = idx >> 6;         // 0..23
    const int chunk = idx & 63;
    const int r = (tile >= 12) ? tile - 12 : tile;
    const int g = r % 3, ct = r / 3;
    const int nt = g * 32 + wg * 4 + ct;
    const _Float16* src = ((tile >= 12) ? Wh : Wi) + (size_t)(nt * KT_H + kt) * 512 + chunk * 8;
    _Float16* d = dst + tile * 512 + chunk * 8;
#ifdef HAVE_ASYNC_LDS
    v4i* s0 = (v4i*)src;   // reinterpret + drop const (same addrspace)
    v4i* d0 = (v4i*)d;
    __builtin_amdgcn_global_load_async_to_lds_b128((gas_v4ip)s0, (las_v4ip)d0, 0, 0);
#else
    *(v8h*)d = *(const v8h*)src;
#endif
  }
}

__device__ __forceinline__ void wait_stage() {
#ifdef HAVE_ASYNC_LDS
  __builtin_amdgcn_s_wait_asynccnt(0);
#endif
  __syncthreads();
}

// ---------------- GRU layer (templated so A-load provenance is LDS-only or global-only) ----------------

template <bool FIRST>
__device__ __forceinline__ void gru_layer(const _Float16* xshL,               // LDS x_t  (FIRST)
                                          const _Float16* __restrict__ inpG,  // global outs (!FIRST)
                                          const _Float16* __restrict__ hA,
                                          const _Float16* __restrict__ Wi,
                                          const _Float16* __restrict__ Wh,
                                          _Float16* wst,                      // LDS weight staging (2 bufs)
                                          const float* __restrict__ bihl,
                                          const float* __restrict__ bhhl,
                                          _Float16* __restrict__ ol,
                                          int wg, int wave, int lane, int tid) {
  const int lh  = lane >> 4;
  const int m0  = (wave >> 1) * 16;     // waves 2k,2k+1 share the M-tile -> one af/hf load feeds 12 WMMAs
  const int ct0 = (wave & 1) * 2;

  stage_wtiles(Wi, Wh, wst, wg, 0, tid);
  wait_stage();

  v8f AI[2][3], AH[2][3];
#pragma unroll
  for (int c = 0; c < 2; ++c)
#pragma unroll
    for (int g = 0; g < 3; ++g) { AI[c][g] = vzero8(); AH[c][g] = vzero8(); }

  for (int kt = 0; kt < KT_H; ++kt) {
    const _Float16* cur = wst + (size_t)(kt & 1) * WBUF_HALFS;
    if (kt + 1 < KT_H)
      stage_wtiles(Wi, Wh, wst + (size_t)((kt + 1) & 1) * WBUF_HALFS, wg, kt + 1, tid);

    v16h af = FIRST ? load_afrag(xshL, IN_, m0, kt * 32, lane)
                    : load_afrag(inpG, H_, m0, kt * 32, lane);
    v16h hf = load_afrag(hA, H_, m0, kt * 32, lane);
#pragma unroll
    for (int c = 0; c < 2; ++c) {
      const int ct = ct0 + c;
#pragma unroll
      for (int g = 0; g < 3; ++g) {
        v16h bi = *(const v16h*)(cur + (ct * 3 + g) * 512 + lane * 16);
        v16h bh = *(const v16h*)(cur + (12 + ct * 3 + g) * 512 + lane * 16);
        AI[c][g] = wmma16(af, bi, AI[c][g]);
        AH[c][g] = wmma16(hf, bh, AH[c][g]);
      }
    }
    wait_stage();   // buf[cur] fully consumed by all waves; buf[nxt] staged for next kt
  }

  // gate math (fp32): r,z,n then h' = (1-z)*n + z*h
#pragma unroll
  for (int c = 0; c < 2; ++c) {
    const int col = wg * HS + (ct0 + c) * 16 + (lane & 15);
    const float bir = bihl[col],          bhr = bhhl[col];
    const float biz = bihl[H_ + col],     bhz = bhhl[H_ + col];
    const float bin = bihl[2 * H_ + col], bhn = bhhl[2 * H_ + col];
#pragma unroll
    for (int v = 0; v < 8; ++v) {
      const int m = m0 + v + 8 * lh;
      const float hp = (float)hA[(size_t)m * H_ + col];
      const float r  = sigmoidf_(AI[c][0][v] + AH[c][0][v] + bir + bhr);
      const float z  = sigmoidf_(AI[c][1][v] + AH[c][1][v] + biz + bhz);
      const float nn = tanhf(AI[c][2][v] + bin + r * (AH[c][2][v] + bhn));
      ol[(size_t)m * H_ + col] = (_Float16)((1.0f - z) * nn + z * hp);
    }
  }
}

// ---------------- prep kernels ----------------

// swizzle fp32 weight W[N][K] (used as B[k][n]=W[n][k]) into WMMA B-fragment order:
// dst[((nt*KT + kt)*32 + lane)*16 + e] ; n = nt*16 + (lane&15) ; k = kt*32 + (e>>3)*16 + (lane>>4)*8 + (e&7)
__global__ void swizzle_w(const float* __restrict__ src, _Float16* __restrict__ dst, int N, int K) {
  int gid = blockIdx.x * blockDim.x + threadIdx.x;
  if (gid >= N * K) return;
  int e    = gid & 15;
  int lane = (gid >> 4) & 31;
  int rest = gid >> 9;
  int KT   = K / 32;
  int kt   = rest % KT;
  int nt   = rest / KT;
  int n = nt * 16 + (lane & 15);
  int k = kt * 32 + (e >> 3) * 16 + (lane >> 4) * 8 + (e & 7);
  dst[gid] = (_Float16)src[(size_t)n * K + k];
}

__global__ void init_state(_Float16* h16, unsigned* cnt) {
  int gid = blockIdx.x * blockDim.x + threadIdx.x;
  if (gid == 0) *cnt = 0u;
  const int n = L_ * B_ * H_;
  for (int i = gid; i < n; i += gridDim.x * blockDim.x) h16[i] = (_Float16)0.0f;
}

// ---------------- persistent HRNN kernel ----------------

__global__ __launch_bounds__(256, 1)
void hrnn_persistent(const float* __restrict__ x,
                     const float* __restrict__ bih, const float* __restrict__ bhh,
                     const float* __restrict__ bm1, const float* __restrict__ bm2,
                     const float* __restrict__ Wm3, const float* __restrict__ bm3,
                     const _Float16* __restrict__ WihS, const _Float16* __restrict__ WhhS,
                     const _Float16* __restrict__ Wm1S, const _Float16* __restrict__ Wm2S,
                     _Float16* h16, _Float16* outs16, _Float16* h1_16, _Float16* h2_16,
                     float* probs, unsigned* barrier_cnt, float* dout) {
  extern __shared__ __align__(16) char smem_raw[];
  _Float16* xsh = (_Float16*)smem_raw;                              // 64KB: x_t as f16
  _Float16* wst = (_Float16*)(smem_raw + (size_t)B_ * IN_ * 2);     // 48KB: weight double-buffer

  const int lane = threadIdx.x & 31;
  const int wave = threadIdx.x >> 5;      // 8 waves
  const int wg   = blockIdx.x;            // 0..7, owns H columns [wg*64, wg*64+64)
  const int lh   = lane >> 4;
  const int tid  = threadIdx.x;
  unsigned gen = 0;

  for (int t = 0; t < T_; ++t) {
    // ---- stage x_t into LDS as f16 ----
    for (int i = tid; i < B_ * IN_; i += 256) {
      int b = i >> 9, k = i & (IN_ - 1);
      xsh[i] = (_Float16)x[((size_t)b * T_ + t) * IN_ + k];
    }
    __syncthreads();

    // ---- 3 stacked GRU layers (sequential; column-parallel across WGs) ----
    for (int l = 0; l < L_; ++l) {
      const _Float16* hA = h16 + (size_t)l * B_ * H_;
      const _Float16* Wi = WihS + (size_t)l * NT_G * KT_H * 512;
      const _Float16* Wh = WhhS + (size_t)l * NT_G * KT_H * 512;
      _Float16* ol = outs16 + (size_t)l * B_ * H_;
      if (l == 0)
        gru_layer<true>(xsh, (const _Float16*)nullptr, hA, Wi, Wh, wst,
                        bih, bhh, ol, wg, wave, lane, tid);
      else
        gru_layer<false>((const _Float16*)nullptr, outs16 + (size_t)(l - 1) * B_ * H_,
                         hA, Wi, Wh, wst,
                         bih + (size_t)l * 3 * H_, bhh + (size_t)l * 3 * H_,
                         ol, wg, wave, lane, tid);
      gbar(barrier_cnt, gen);
    }

    // ---- transition MLP layer 1: h1 = relu(out_l @ Wm1^T + bm1), both transitions ----
    for (int it = 0; it < 4; ++it) {             // 32 tiles / 8 waves
      const int idx = wave * 4 + it;
      const int tl = idx >> 4;
      const int tile = idx & 15;
      const int m0 = (tile >> 2) * 16;
      const int c0 = wg * HS + (tile & 3) * 16;
      const _Float16* Aa = outs16 + (size_t)tl * B_ * H_;
      const _Float16* Ws = Wm1S + (size_t)tl * NT_H * KT_H * 512;
      v8f acc = vzero8();
      for (int kt = 0; kt < KT_H; ++kt) {
        if (kt + 1 < KT_H)
          __builtin_prefetch(Ws + (((size_t)(c0 >> 4) * KT_H + kt + 1) * 32) * 16, 0, 1);
        acc = wmma16(load_afrag(Aa, H_, m0, kt * 32, lane),
                     load_bfrag(Ws, KT_H, c0 >> 4, kt, lane), acc);
      }
      const int col = c0 + (lane & 15);
      const float bb = bm1[tl * H_ + col];
      for (int v = 0; v < 8; ++v) {
        const float val = acc[v] + bb;
        h1_16[(size_t)tl * B_ * H_ + (size_t)(m0 + v + 8 * lh) * H_ + col] =
            (_Float16)(val > 0.0f ? val : 0.0f);
      }
    }
    gbar(barrier_cnt, gen);

    // ---- transition MLP layer 2: h2 = relu(h1 @ Wm2^T + bm2) ----
    for (int it = 0; it < 4; ++it) {
      const int idx = wave * 4 + it;
      const int tl = idx >> 4;
      const int tile = idx & 15;
      const int m0 = (tile >> 2) * 16;
      const int c0 = wg * HS + (tile & 3) * 16;
      const _Float16* Aa = h1_16 + (size_t)tl * B_ * H_;
      const _Float16* Ws = Wm2S + (size_t)tl * NT_H * KT_H * 512;
      v8f acc = vzero8();
      for (int kt = 0; kt < KT_H; ++kt) {
        if (kt + 1 < KT_H)
          __builtin_prefetch(Ws + (((size_t)(c0 >> 4) * KT_H + kt + 1) * 32) * 16, 0, 1);
        acc = wmma16(load_afrag(Aa, H_, m0, kt * 32, lane),
                     load_bfrag(Ws, KT_H, c0 >> 4, kt, lane), acc);
      }
      const int col = c0 + (lane & 15);
      const float bb = bm2[tl * H_ + col];
      for (int v = 0; v < 8; ++v) {
        const float val = acc[v] + bb;
        h2_16[(size_t)tl * B_ * H_ + (size_t)(m0 + v + 8 * lh) * H_ + col] =
            (_Float16)(val > 0.0f ? val : 0.0f);
      }
    }
    gbar(barrier_cnt, gen);

    // ---- transition probs: p = sigmoid(h2 @ Wm3^T + bm3), write boundary output ----
    if (wg < L_ - 1) {
      const int tl = wg;
      for (int b = tid; b < B_; b += 256) {
        float acc = bm3[tl];
        const _Float16* hr = h2_16 + (size_t)tl * B_ * H_ + (size_t)b * H_;
        const float* w3 = Wm3 + (size_t)tl * H_;
        for (int k = 0; k < H_; ++k) acc += (float)hr[k] * w3[k];
        const float p = sigmoidf_(acc);
        probs[tl * B_ + b] = p;
        dout[((size_t)b * T_ + t) * (L_ - 1) + tl] = p;   // [B, T, L-1]
      }
    }
    gbar(barrier_cnt, gen);

    // ---- hierarchical state mixing (each WG updates its own H-slice, all layers) ----
    for (int i = tid; i < B_ * HS; i += 256) {
      const int b = i / HS;
      const int col = wg * HS + (i % HS);
      const float p0 = probs[b], p1 = probs[B_ + b];
      const size_t o = (size_t)b * H_ + col;
      const float o0 = (float)outs16[o];
      const float o1 = (float)outs16[(size_t)B_ * H_ + o];
      const float o2 = (float)outs16[(size_t)2 * B_ * H_ + o];
      const float m1v = (1.0f - p1) * o1 + p1 * o2;
      h16[o] = (_Float16)((1.0f - p0) * o0 + p0 * m1v);
      h16[(size_t)B_ * H_ + o] = (_Float16)m1v;
      h16[(size_t)2 * B_ * H_ + o] = (_Float16)o2;
    }
    gbar(barrier_cnt, gen);
  }
}

// ---------------- host launch ----------------

extern "C" void kernel_launch(void* const* d_in, const int* in_sizes, int n_in,
                              void* d_out, int out_size, void* d_ws, size_t ws_size,
                              hipStream_t stream) {
  const float* x   = (const float*)d_in[0];
  const float* Wih = (const float*)d_in[1];
  const float* Whh = (const float*)d_in[2];
  const float* bih = (const float*)d_in[3];
  const float* bhh = (const float*)d_in[4];
  const float* Wm1 = (const float*)d_in[5];
  const float* bm1 = (const float*)d_in[6];
  const float* Wm2 = (const float*)d_in[7];
  const float* bm2 = (const float*)d_in[8];
  const float* Wm3 = (const float*)d_in[9];
  const float* bm3 = (const float*)d_in[10];
  float* dout = (float*)d_out;

  char* ws = (char*)d_ws;
  size_t off = 0;
  auto alloc = [&](size_t bytes) -> void* {
    void* p = ws + off;
    off = (off + bytes + 255) & ~(size_t)255;
    return p;
  };
  _Float16* WihS = (_Float16*)alloc((size_t)L_ * 3 * H_ * IN_ * sizeof(_Float16));
  _Float16* WhhS = (_Float16*)alloc((size_t)L_ * 3 * H_ * H_ * sizeof(_Float16));
  _Float16* Wm1S = (_Float16*)alloc((size_t)(L_ - 1) * H_ * H_ * sizeof(_Float16));
  _Float16* Wm2S = (_Float16*)alloc((size_t)(L_ - 1) * H_ * H_ * sizeof(_Float16));
  _Float16* h16    = (_Float16*)alloc((size_t)L_ * B_ * H_ * sizeof(_Float16));
  _Float16* outs16 = (_Float16*)alloc((size_t)L_ * B_ * H_ * sizeof(_Float16));
  _Float16* h1_16  = (_Float16*)alloc((size_t)(L_ - 1) * B_ * H_ * sizeof(_Float16));
  _Float16* h2_16  = (_Float16*)alloc((size_t)(L_ - 1) * B_ * H_ * sizeof(_Float16));
  float*    probs  = (float*)alloc((size_t)(L_ - 1) * B_ * sizeof(float));
  unsigned* cnt    = (unsigned*)alloc(256);

  // pre-swizzle weights to f16 WMMA B-fragment layout
  for (int l = 0; l < L_; ++l) {
    const int tg = 3 * H_ * IN_;
    swizzle_w<<<(tg + 255) / 256, 256, 0, stream>>>(Wih + (size_t)l * tg, WihS + (size_t)l * tg, 3 * H_, IN_);
    swizzle_w<<<(tg + 255) / 256, 256, 0, stream>>>(Whh + (size_t)l * tg, WhhS + (size_t)l * tg, 3 * H_, H_);
  }
  for (int l = 0; l < L_ - 1; ++l) {
    const int th = H_ * H_;
    swizzle_w<<<(th + 255) / 256, 256, 0, stream>>>(Wm1 + (size_t)l * th, Wm1S + (size_t)l * th, H_, H_);
    swizzle_w<<<(th + 255) / 256, 256, 0, stream>>>(Wm2 + (size_t)l * th, Wm2S + (size_t)l * th, H_, H_);
  }
  // zero initial hidden state + barrier counter (every call -> deterministic replay)
  init_state<<<64, 256, 0, stream>>>(h16, cnt);

  hrnn_persistent<<<NWG, 256, SMEM_BYTES, stream>>>(x, bih, bhh, bm1, bm2, Wm3, bm3,
                                                    WihS, WhhS, Wm1S, Wm2S,
                                                    h16, outs16, h1_16, h2_16,
                                                    probs, cnt, dout);
}
